// DecoderLSTM_69621419868618
// MI455X (gfx1250) — matrix operand order repaired and checked by reading, not compiled, not run on hardware
//
#include <hip/hip_runtime.h>
#include <hip/hip_bf16.h>

#define B_    128
#define S_    2048
#define H_    512
#define OUT_  32

typedef __attribute__((ext_vector_type(16))) _Float16 v16h;
typedef __attribute__((ext_vector_type(8)))  float    v8f;
typedef __attribute__((ext_vector_type(4)))  int      v4i;

union HV { v16h v; _Float16 e[16]; uint4 q[2]; };

typedef __attribute__((address_space(1))) v4i* gv4i_ptr;   // global int4*
typedef __attribute__((address_space(3))) v4i* lv4i_ptr;   // LDS int4*

__device__ __forceinline__ float tanh_fast(float x) {
    float t2 = __expf(2.0f * x);
    return 1.0f - 2.0f / (t2 + 1.0f);          // robust at +/- inf
}
__device__ __forceinline__ float sigmoid_fast(float x) {
    return 1.0f / (1.0f + __expf(-x));
}
__device__ __forceinline__ float dot512(const float* __restrict__ a,
                                        const float* __restrict__ w) {
    float acc = 0.f;
    #pragma unroll 4
    for (int k = 0; k < 512; k += 4) {
        float4 av = *(const float4*)(a + k);
        float4 wv = *(const float4*)(w + k);
        acc += av.x * wv.x + av.y * wv.y + av.z * wv.z + av.w * wv.w;
    }
    return acc;
}

__device__ __forceinline__ void wait_async0() {
#if __has_builtin(__builtin_amdgcn_s_wait_asynccnt)
    __builtin_amdgcn_s_wait_asynccnt(0);
#else
    asm volatile("s_wait_asynccnt 0x0" ::: "memory");
#endif
}

// Stage 128 bytes (8 x b128) from global to LDS, per thread.
// Prefers the CDNA5 async DMA path (no data VGPRs, ASYNCcnt-tracked).
__device__ __forceinline__ void stage128(const _Float16* __restrict__ gsrc,
                                         _Float16* lds_dst) {
#if __has_builtin(__builtin_amdgcn_global_load_async_to_lds_b128)
    gv4i_ptr g = (gv4i_ptr)(unsigned long long)(const void*)gsrc;
    lv4i_ptr l = (lv4i_ptr)(unsigned)(unsigned long long)(void*)lds_dst;
    #pragma unroll
    for (int j = 0; j < 8; ++j)
        __builtin_amdgcn_global_load_async_to_lds_b128(g + j, l + j, 0, 0);
#else
    const uint4* src = (const uint4*)gsrc;
    uint4* dst = (uint4*)lds_dst;
    #pragma unroll
    for (int j = 0; j < 8; ++j) dst[j] = src[j];
#endif
}

// ---------------------------------------------------------------------------
// 1) W_e (attn_W[:, 512:]) -> f16, row-major [512 x 512]
__global__ __launch_bounds__(256) void prep_We(const float* __restrict__ attnW,
                                               _Float16* __restrict__ Weh) {
    int id = blockIdx.x * 256 + threadIdx.x;   // 262144
    int h = id >> 9, k = id & 511;
    Weh[id] = (_Float16)attnW[h * 1024 + 512 + k];
}

// 2) biasH[b,h] = hid(=h0[1]) . W_h[h,:] + attn_b[h]
__global__ __launch_bounds__(256) void bias_h_kernel(const float* __restrict__ h0,
                                                     const float* __restrict__ attnW,
                                                     const float* __restrict__ attnB,
                                                     float* __restrict__ biasH) {
    int id = blockIdx.x * 256 + threadIdx.x;   // 65536
    int b = id >> 9, h = id & 511;
    const float* hid = h0 + (size_t)B_ * H_ + (size_t)b * H_;   // layer -1
    biasH[id] = dot512(hid, attnW + (size_t)h * 1024) + attnB[h];
}

// 3) scores[b,s] = sum_h v[h]*tanh(enc[b,s,:]@We[h,:] + biasH[b,h])
//    Block = 4 waves = 128 bs rows (same b). Each wave owns two 16-row A tiles
//    (256 VGPRs of f16 A). B tile (16 x 512 f16) is staged into double-buffered
//    LDS via async global->LDS DMA (overlapped with WMMA of the previous tile)
//    and shared by all 4 waves -> 8x B reuse per L2 read.
#define BPAD 520   // padded row stride in f16 (bank-conflict-free b128 reads)
__global__ __launch_bounds__(128) void attn_scores_wmma(const float* __restrict__ enc,
                                                        const _Float16* __restrict__ Weh,
                                                        const float* __restrict__ biasH,
                                                        const float* __restrict__ vvec,
                                                        float* __restrict__ scores) {
    __shared__ _Float16 Bsh[2][16 * BPAD];     // 2 x 16.6 KB

    const int lane = threadIdx.x & 31;
    const int wave = threadIdx.x >> 5;
    const long bs0b = (long)blockIdx.x * 128;  // block's first bs row
    const long bs0w = bs0b + wave * 32;        // wave's first bs row (2 tiles)
    const int  b    = (int)(bs0b >> 11);       // S=2048; block stays in one b
    const int  m    = lane & 15;
    const int  half = lane >> 4;

    const int tr = threadIdx.x >> 3;           // staging: row 0..15
    const int tc = (threadIdx.x & 7) * 64;     // staging: 64 f16 per thread

    // kick off stage of N-tile 0 into buffer 0 before the A preload
    stage128(Weh + (size_t)tr * 512 + tc, &Bsh[0][tr * BPAD + tc]);

    // Preload A: 2 tiles x 16 rows x 512 K in f16 (ISA 16-bit A layout)
    HV a0[16], a1[16];
    const float* arow0 = enc + (bs0w + m) * (long)H_;
    const float* arow1 = enc + (bs0w + 16 + m) * (long)H_;
    #pragma unroll
    for (int kt = 0; kt < 16; ++kt) {
        int c0 = kt * 32 + half * 8;
        float4 f0 = *(const float4*)(arow0 + c0);
        float4 f1 = *(const float4*)(arow0 + c0 + 4);
        float4 f2 = *(const float4*)(arow0 + c0 + 16);
        float4 f3 = *(const float4*)(arow0 + c0 + 20);
        a0[kt].e[0]=(_Float16)f0.x;  a0[kt].e[1]=(_Float16)f0.y;
        a0[kt].e[2]=(_Float16)f0.z;  a0[kt].e[3]=(_Float16)f0.w;
        a0[kt].e[4]=(_Float16)f1.x;  a0[kt].e[5]=(_Float16)f1.y;
        a0[kt].e[6]=(_Float16)f1.z;  a0[kt].e[7]=(_Float16)f1.w;
        a0[kt].e[8]=(_Float16)f2.x;  a0[kt].e[9]=(_Float16)f2.y;
        a0[kt].e[10]=(_Float16)f2.z; a0[kt].e[11]=(_Float16)f2.w;
        a0[kt].e[12]=(_Float16)f3.x; a0[kt].e[13]=(_Float16)f3.y;
        a0[kt].e[14]=(_Float16)f3.z; a0[kt].e[15]=(_Float16)f3.w;
        float4 g0 = *(const float4*)(arow1 + c0);
        float4 g1 = *(const float4*)(arow1 + c0 + 4);
        float4 g2 = *(const float4*)(arow1 + c0 + 16);
        float4 g3 = *(const float4*)(arow1 + c0 + 20);
        a1[kt].e[0]=(_Float16)g0.x;  a1[kt].e[1]=(_Float16)g0.y;
        a1[kt].e[2]=(_Float16)g0.z;  a1[kt].e[3]=(_Float16)g0.w;
        a1[kt].e[4]=(_Float16)g1.x;  a1[kt].e[5]=(_Float16)g1.y;
        a1[kt].e[6]=(_Float16)g1.z;  a1[kt].e[7]=(_Float16)g1.w;
        a1[kt].e[8]=(_Float16)g2.x;  a1[kt].e[9]=(_Float16)g2.y;
        a1[kt].e[10]=(_Float16)g2.z; a1[kt].e[11]=(_Float16)g2.w;
        a1[kt].e[12]=(_Float16)g3.x; a1[kt].e[13]=(_Float16)g3.y;
        a1[kt].e[14]=(_Float16)g3.z; a1[kt].e[15]=(_Float16)g3.w;
    }

    float sacc0[8], sacc1[8];
    #pragma unroll
    for (int r = 0; r < 8; ++r) { sacc0[r] = 0.f; sacc1[r] = 0.f; }

    wait_async0();
    __syncthreads();                            // tile 0 staged

    for (int nt = 0; nt < 32; ++nt) {
        const int cur = nt & 1;
        const int n0  = nt * 16;
        // overlap: stage next N-tile into the other buffer while computing
        if (nt < 31) {
            stage128(Weh + (size_t)(n0 + 16 + tr) * 512 + tc,
                     &Bsh[cur ^ 1][tr * BPAD + tc]);
        }

        const float bias = biasH[b * H_ + n0 + m];
        const float vv   = vvec[n0 + m];
        v8f acc0 = {}, acc1 = {};
        const _Float16* bp = &Bsh[cur][m * BPAD + half * 8];

        // double-buffered B registers: load bb[kt+1] before consuming bb[kt]
        HV bbA, bbB;
        bbA.q[0] = *(const uint4*)(bp);
        bbA.q[1] = *(const uint4*)(bp + 16);
        #pragma unroll
        for (int kt = 0; kt < 16; kt += 2) {
            bbB.q[0] = *(const uint4*)(bp + (kt + 1) * 32);
            bbB.q[1] = *(const uint4*)(bp + (kt + 1) * 32 + 16);
            acc0 = __builtin_amdgcn_wmma_f32_16x16x32_f16(
                       false, a0[kt].v, false, bbA.v, (short)0, acc0, false, false);
            acc1 = __builtin_amdgcn_wmma_f32_16x16x32_f16(
                       false, a1[kt].v, false, bbA.v, (short)0, acc1, false, false);
            if (kt + 2 < 16) {
                bbA.q[0] = *(const uint4*)(bp + (kt + 2) * 32);
                bbA.q[1] = *(const uint4*)(bp + (kt + 2) * 32 + 16);
            }
            acc0 = __builtin_amdgcn_wmma_f32_16x16x32_f16(
                       false, a0[kt + 1].v, false, bbB.v, (short)0, acc0, false, false);
            acc1 = __builtin_amdgcn_wmma_f32_16x16x32_f16(
                       false, a1[kt + 1].v, false, bbB.v, (short)0, acc1, false, false);
        }
        #pragma unroll
        for (int r = 0; r < 8; ++r) {
            sacc0[r] += tanh_fast(acc0[r] + bias) * vv;
            sacc1[r] += tanh_fast(acc1[r] + bias) * vv;
        }
        wait_async0();                          // next tile's DMA complete
        __syncthreads();                        // all waves done with cur
    }

    // reduce the 16 columns held by each half-wave
    #pragma unroll
    for (int r = 0; r < 8; ++r) {
        float s0 = sacc0[r], s1 = sacc1[r];
        s0 += __shfl_xor(s0, 1, 32); s1 += __shfl_xor(s1, 1, 32);
        s0 += __shfl_xor(s0, 2, 32); s1 += __shfl_xor(s1, 2, 32);
        s0 += __shfl_xor(s0, 4, 32); s1 += __shfl_xor(s1, 4, 32);
        s0 += __shfl_xor(s0, 8, 32); s1 += __shfl_xor(s1, 8, 32);
        sacc0[r] = s0; sacc1[r] = s1;
    }
    if (m == 0) {                               // lanes 0 and 16
        float* srow0 = scores + bs0w + half * 8;
        float* srow1 = scores + bs0w + 16 + half * 8;
        #pragma unroll
        for (int r = 0; r < 8; ++r) { srow0[r] = sacc0[r]; srow1[r] = sacc1[r]; }
    }
}

// 4) softmax over S per batch row -> attn_weights (into d_out)
__global__ __launch_bounds__(256) void softmax_rows(const float* __restrict__ scores,
                                                    float* __restrict__ attnw) {
    int b = blockIdx.x;
    __shared__ float red[256];
    const float* row = scores + (size_t)b * S_;
    float mx = -1e30f;
    for (int s = threadIdx.x; s < S_; s += 256) mx = fmaxf(mx, row[s]);
    red[threadIdx.x] = mx; __syncthreads();
    for (int st = 128; st > 0; st >>= 1) {
        if (threadIdx.x < st) red[threadIdx.x] = fmaxf(red[threadIdx.x], red[threadIdx.x + st]);
        __syncthreads();
    }
    mx = red[0]; __syncthreads();
    float sum = 0.f;
    for (int s = threadIdx.x; s < S_; s += 256) sum += __expf(row[s] - mx);
    red[threadIdx.x] = sum; __syncthreads();
    for (int st = 128; st > 0; st >>= 1) {
        if (threadIdx.x < st) red[threadIdx.x] += red[threadIdx.x + st];
        __syncthreads();
    }
    sum = red[0];
    float inv = 1.0f / sum;
    for (int s = threadIdx.x; s < S_; s += 256)
        attnw[(size_t)b * S_ + s] = __expf(row[s] - mx) * inv;
}

// 5a) context partial sums (split-S, deterministic: no atomics)
__global__ __launch_bounds__(256) void context_partial(const float* __restrict__ enc,
                                                       const float* __restrict__ attnw,
                                                       float* __restrict__ partial) {
    int chunk = blockIdx.x;                     // 8 chunks of 256 s
    int b     = blockIdx.y;
    int t     = threadIdx.x;
    __shared__ float w[256];
    int sbase = chunk * 256;
    w[t] = attnw[(size_t)b * S_ + sbase + t];
    __syncthreads();
    float a0 = 0.f, a1 = 0.f;
    const float* e = enc + ((long)b * S_ + sbase) * H_;
    for (int s = 0; s < 256; ++s) {
        float ws = w[s];
        a0 += ws * e[(long)s * H_ + t];
        a1 += ws * e[(long)s * H_ + t + 256];
    }
    float* p = partial + ((size_t)chunk * B_ + b) * H_;
    p[t] = a0; p[t + 256] = a1;
}

// 5b) reduce 8 chunks -> context[b,h]
__global__ __launch_bounds__(256) void context_reduce(const float* __restrict__ partial,
                                                      float* __restrict__ ctx) {
    int id = blockIdx.x * 256 + threadIdx.x;    // 65536
    float a = 0.f;
    #pragma unroll
    for (int c = 0; c < 8; ++c) a += partial[(size_t)c * B_ * H_ + id];
    ctx[id] = a;
}

// 6) LSTM cell layer 0: x_step = [x(32) | context(512)]
__global__ __launch_bounds__(256) void lstm_cell0(const float* __restrict__ x,
                                                  const float* __restrict__ ctx,
                                                  const float* __restrict__ h0,
                                                  const float* __restrict__ c0,
                                                  const float* __restrict__ Wih,
                                                  const float* __restrict__ Whh,
                                                  const float* __restrict__ bih,
                                                  const float* __restrict__ bhh,
                                                  float* __restrict__ h1ws,
                                                  float* __restrict__ outHN,
                                                  float* __restrict__ outCN) {
    int id = blockIdx.x * 256 + threadIdx.x;    // 65536
    int b = id >> 9, h = id & 511;
    const float* xb = x + b * OUT_;
    const float* cb = ctx + (size_t)b * H_;
    const float* hb = h0 + (size_t)b * H_;      // layer 0
    float g[4];
    #pragma unroll
    for (int gi = 0; gi < 4; ++gi) {
        int row = gi * H_ + h;
        const float* wi = Wih + (size_t)row * 544;
        float acc = bih[row] + bhh[row];
        #pragma unroll
        for (int k = 0; k < 32; k += 4) {
            float4 av = *(const float4*)(xb + k);
            float4 wv = *(const float4*)(wi + k);
            acc += av.x * wv.x + av.y * wv.y + av.z * wv.z + av.w * wv.w;
        }
        acc += dot512(cb, wi + 32);
        acc += dot512(hb, Whh + (size_t)row * 512);
        g[gi] = acc;
    }
    float i = sigmoid_fast(g[0]);
    float f = sigmoid_fast(g[1]);
    float gg = tanh_fast(g[2]);
    float o = sigmoid_fast(g[3]);
    float cn = f * c0[(size_t)b * H_ + h] + i * gg;
    float hn = o * tanh_fast(cn);
    h1ws[id] = hn;
    outHN[id] = hn;          // h_new[0]
    outCN[id] = cn;          // c_new[0]
}

// 7) LSTM cell layer 1
__global__ __launch_bounds__(256) void lstm_cell1(const float* __restrict__ h1,
                                                  const float* __restrict__ h0l1,
                                                  const float* __restrict__ c0l1,
                                                  const float* __restrict__ Wih,
                                                  const float* __restrict__ Whh,
                                                  const float* __restrict__ bih,
                                                  const float* __restrict__ bhh,
                                                  float* __restrict__ h2ws,
                                                  float* __restrict__ outHN,
                                                  float* __restrict__ outCN) {
    int id = blockIdx.x * 256 + threadIdx.x;    // 65536
    int b = id >> 9, h = id & 511;
    const float* xb = h1 + (size_t)b * H_;
    const float* hb = h0l1 + (size_t)b * H_;
    float g[4];
    #pragma unroll
    for (int gi = 0; gi < 4; ++gi) {
        int row = gi * H_ + h;
        float acc = bih[row] + bhh[row];
        acc += dot512(xb, Wih + (size_t)row * 512);
        acc += dot512(hb, Whh + (size_t)row * 512);
        g[gi] = acc;
    }
    float i = sigmoid_fast(g[0]);
    float f = sigmoid_fast(g[1]);
    float gg = tanh_fast(g[2]);
    float o = sigmoid_fast(g[3]);
    float cn = f * c0l1[(size_t)b * H_ + h] + i * gg;
    float hn = o * tanh_fast(cn);
    h2ws[id] = hn;
    outHN[id] = hn;          // h_new[1]
    outCN[id] = cn;          // c_new[1]
}

// 8) projection: out[b,o] = h2 . proj_W[o,:] + proj_b[o]
__global__ __launch_bounds__(256) void proj_kernel(const float* __restrict__ h2,
                                                   const float* __restrict__ pW,
                                                   const float* __restrict__ pb,
                                                   float* __restrict__ out) {
    int id = blockIdx.x * 256 + threadIdx.x;    // 4096
    int b = id >> 5, o = id & 31;
    out[id] = dot512(h2 + (size_t)b * H_, pW + (size_t)o * H_) + pb[o];
}

extern "C" void kernel_launch(void* const* d_in, const int* in_sizes, int n_in,
                              void* d_out, int out_size, void* d_ws, size_t ws_size,
                              hipStream_t stream) {
    const float* x     = (const float*)d_in[0];
    const float* h0    = (const float*)d_in[1];
    const float* c0    = (const float*)d_in[2];
    const float* enc   = (const float*)d_in[3];
    const float* attnW = (const float*)d_in[4];
    const float* attnB = (const float*)d_in[5];
    const float* vvec  = (const float*)d_in[6];
    const float* Wih0  = (const float*)d_in[7];
    const float* Whh0  = (const float*)d_in[8];
    const float* bih0  = (const float*)d_in[9];
    const float* bhh0  = (const float*)d_in[10];
    const float* Wih1  = (const float*)d_in[11];
    const float* Whh1  = (const float*)d_in[12];
    const float* bih1  = (const float*)d_in[13];
    const float* bhh1  = (const float*)d_in[14];
    const float* projW = (const float*)d_in[15];
    const float* projB = (const float*)d_in[16];

    char* ws = (char*)d_ws;
    _Float16* Weh   = (_Float16*)(ws + 0);            // 512 KB
    float* biasH    = (float*)(ws + 524288);          // 256 KB
    float* scores   = (float*)(ws + 786432);          // 1 MB
    float* partial  = (float*)(ws + 1835008);         // 2 MB
    float* ctx      = (float*)(ws + 3932160);         // 256 KB
    float* h1ws     = (float*)(ws + 4194304);         // 256 KB
    float* h2ws     = (float*)(ws + 4456448);         // 256 KB

    float* out_y  = (float*)d_out;                    // (128,1,32)
    float* out_hn = out_y + 4096;                     // (2,128,512)
    float* out_cn = out_hn + 131072;                  // (2,128,512)
    float* out_aw = out_cn + 131072;                  // (128,2048)

    prep_We<<<1024, 256, 0, stream>>>(attnW, Weh);
    bias_h_kernel<<<256, 256, 0, stream>>>(h0, attnW, attnB, biasH);
    attn_scores_wmma<<<2048, 128, 0, stream>>>(enc, Weh, biasH, vvec, scores);
    softmax_rows<<<128, 256, 0, stream>>>(scores, out_aw);
    context_partial<<<dim3(8, 128), 256, 0, stream>>>(enc, out_aw, partial);
    context_reduce<<<256, 256, 0, stream>>>(partial, ctx);
    lstm_cell0<<<256, 256, 0, stream>>>(x, ctx, h0, c0, Wih0, Whh0, bih0, bhh0,
                                        h1ws, out_hn, out_cn);
    lstm_cell1<<<256, 256, 0, stream>>>(h1ws, h0 + 65536, c0 + 65536,
                                        Wih1, Whh1, bih1, bhh1,
                                        h2ws, out_hn + 65536, out_cn + 65536);
    proj_kernel<<<16, 256, 0, stream>>>(h2ws, projW, projB, out_y);
}
